// PointTransformerBlock_523986010297
// MI455X (gfx1250) — matrix operand order repaired and checked by reading, not compiled, not run on hardware
//
#include <hip/hip_runtime.h>
#include <hip/hip_bf16.h>

// ---------------------------------------------------------------------------
// PointTransformerBlock for MI455X (gfx1250): wave32, WMMA f32<=f16 for the
// five dense 64x64 GEMMs (weights pre-swizzled to per-lane B-fragments so
// each lane loads 2x b128), LDS+float-atomic reductions for training-mode BN
// statistics; pe / w0 recomputed on the fly to avoid 330MB intermediates.
// ---------------------------------------------------------------------------

typedef __attribute__((ext_vector_type(16))) _Float16 v16h;
typedef __attribute__((ext_vector_type(8)))  float    v8f;

#define EPSBN 1e-5f

// Stats buffer layout (float offsets inside workspace stats block):
//  bn1  : sum @   0 (64), sq @  64 (64)
//  bnp  : sum @ 128 ( 3), sq @ 136 ( 3)
//  bnw1 : sum @ 192 (64), sq @ 256 (64)
//  bnw2 : sum @ 320 ( 8), sq @ 328 ( 8)
//  bn2  : sum @ 384 (64), sq @ 448 (64)
//  bn3  : sum @ 512 (64), sq @ 576 (64)
#define STATS_FLOATS 1024
#define WFRAG_HALVES 4096   // one 64x64 weight as f16 B-fragments

__global__ __launch_bounds__(256) void zero_stats_k(float* s) {
    for (int i = threadIdx.x; i < STATS_FLOATS; i += blockDim.x) s[i] = 0.0f;
}

// ---------------------------------------------------------------------------
// Convert a 64x64 f32 weight into pre-swizzled f16 WMMA B-fragments:
// out[((nt*2+kt)*32 + lane)*16 + h] = W[(kt*32 + kbase(lane) + h + (h>=8?8:0))*64 + col]
// where kbase = (lane<16)?0:8, col = nt*16 + (lane&15).  (ISA 7.12.2)
// ---------------------------------------------------------------------------
__global__ __launch_bounds__(256) void prep_weight_k(
    const float* __restrict__ W, _Float16* __restrict__ out)
{
    const int o = blockIdx.x * 256 + threadIdx.x;   // 0..4095
    if (o >= WFRAG_HALVES) return;
    const int h    = o & 15;
    const int lane = (o >> 4) & 31;
    const int ktnt = o >> 9;                        // 0..7
    const int kt   = ktnt & 1;
    const int nt   = ktnt >> 1;
    const int kbase = (lane < 16) ? 0 : 8;
    const int col   = nt * 16 + (lane & 15);
    const int k     = kt * 32 + kbase + h + (h >= 8 ? 8 : 0);
    out[o] = (_Float16)W[k * 64 + col];
}

// ---------------------------------------------------------------------------
// GEMM: out[nrows x 64] = in[nrows x 64] @ W[64 x 64] (+ bias), f32 in/out,
// f16 WMMA compute. One wave per 16-row tile; 8 waves / block => 128 rows.
// A loaded as float4 (b128) + packed cvt; B fragments as v16h (2x b128).
// ---------------------------------------------------------------------------
__global__ __launch_bounds__(256) void gemm64_wmma_k(
    const float* __restrict__ in, const _Float16* __restrict__ Wf,
    const float* __restrict__ bias, float* __restrict__ out, int nrows)
{
    const int wave = blockIdx.x * (blockDim.x >> 5) + (threadIdx.x >> 5);
    const int lane = threadIdx.x & 31;
    const int row0 = wave * 16;
    if (row0 >= nrows) return;

    // ---- A fragments (16x32 f16, two K-halves of K=64) ----
    const int m  = lane & 15;
    const int q4 = (lane < 16) ? 0 : 2;             // kbaseA/4 (0 or 8 floats)
    int arow = row0 + m; if (arow >= nrows) arow = nrows - 1;
    const float4* __restrict__ ap = (const float4*)(in + (size_t)arow * 64);

    float af[16], ag[16];
    *(float4*)(af +  0) = ap[q4 +  0];
    *(float4*)(af +  4) = ap[q4 +  1];
    *(float4*)(af +  8) = ap[q4 +  4];
    *(float4*)(af + 12) = ap[q4 +  5];
    *(float4*)(ag +  0) = ap[q4 +  8];
    *(float4*)(ag +  4) = ap[q4 +  9];
    *(float4*)(ag +  8) = ap[q4 + 12];
    *(float4*)(ag + 12) = ap[q4 + 13];

    v16h a0, a1;
    #pragma unroll
    for (int i = 0; i < 16; ++i) {
        a0[i] = (_Float16)af[i];
        a1[i] = (_Float16)ag[i];
    }

    // ---- B fragments (pre-swizzled) + accumulate over 4 column tiles ----
    const v16h* __restrict__ bf = (const v16h*)Wf;

    v8f acc[4];
    #pragma unroll
    for (int t = 0; t < 4; ++t)
        #pragma unroll
        for (int i = 0; i < 8; ++i) acc[t][i] = 0.0f;

    #pragma unroll
    for (int nt = 0; nt < 4; ++nt) {
        const v16h b0 = bf[(nt * 2 + 0) * 32 + lane];
        const v16h b1 = bf[(nt * 2 + 1) * 32 + lane];
        acc[nt] = __builtin_amdgcn_wmma_f32_16x16x32_f16(
            false, a0, false, b0, (short)0, acc[nt], false, false);
        acc[nt] = __builtin_amdgcn_wmma_f32_16x16x32_f16(
            false, a1, false, b1, (short)0, acc[nt], false, false);
    }

    // ---- store C/D: lanes 0-15 -> M=r, lanes 16-31 -> M=8+r; N = lane&15 ----
    const int n     = lane & 15;
    const int mbase = (lane < 16) ? 0 : 8;
    #pragma unroll
    for (int nt = 0; nt < 4; ++nt) {
        const int col = nt * 16 + n;
        const float bs = bias ? bias[col] : 0.0f;
        #pragma unroll
        for (int r = 0; r < 8; ++r) {
            const int row = row0 + mbase + r;
            if (row < nrows) out[(size_t)row * 64 + col] = acc[nt][r] + bs;
        }
    }
}

// ---------------------------------------------------------------------------
// Per-channel (64) sum & sum-of-squares over rows. LDS reduce + global atomics.
// ---------------------------------------------------------------------------
__global__ __launch_bounds__(256) void colstats64_k(
    const float* __restrict__ x, float* __restrict__ osum,
    float* __restrict__ osq, int nrows)
{
    __shared__ float ssum[256], ssq[256];
    const int c = threadIdx.x & 63;
    const int g = threadIdx.x >> 6;
    float s = 0.0f, q = 0.0f;
    for (int r = blockIdx.x * 4 + g; r < nrows; r += gridDim.x * 4) {
        const float v = x[(size_t)r * 64 + c];
        s += v; q += v * v;
    }
    ssum[threadIdx.x] = s; ssq[threadIdx.x] = q;
    __syncthreads();
    if (threadIdx.x < 64) {
        s = ssum[c] + ssum[c + 64] + ssum[c + 128] + ssum[c + 192];
        q = ssq[c]  + ssq[c + 64]  + ssq[c + 128]  + ssq[c + 192];
        atomicAdd(&osum[c], s);
        atomicAdd(&osq[c], q);
    }
}

// out = relu((in - mean) * rsqrt(var+eps) * g + b), channels = i & 63
__global__ __launch_bounds__(256) void bn_relu_k(
    const float* __restrict__ in, const float* __restrict__ sum,
    const float* __restrict__ sq, const float* __restrict__ g,
    const float* __restrict__ b, float* __restrict__ out,
    int total, float invcnt)
{
    const int i = blockIdx.x * blockDim.x + threadIdx.x;
    if (i >= total) return;
    const int c = i & 63;
    const float m  = sum[c] * invcnt;
    const float v  = sq[c] * invcnt - m * m;
    const float y  = (in[i] - m) * rsqrtf(v + EPSBN) * g[c] + b[c];
    out[i] = fmaxf(y, 0.0f);
}

// ---------------------------------------------------------------------------
// pr1[i,j,:] = (p[knn[i,j]] - p[i]) @ Wp1 + bp1  (N*NS x 3), + bnp stats
// ---------------------------------------------------------------------------
__global__ __launch_bounds__(256) void pr_stage_k(
    const float* __restrict__ p, const int* __restrict__ knn,
    const float* __restrict__ Wp1, const float* __restrict__ bp1,
    float* __restrict__ pr1, float* __restrict__ S, int npairs)
{
    __shared__ float sacc[8];
    if (threadIdx.x < 8) sacc[threadIdx.x] = 0.0f;
    __syncthreads();

    const int pair = blockIdx.x * blockDim.x + threadIdx.x;
    if (pair < npairs) {
        const int i  = pair >> 4;
        const int nb = knn[pair];
        const float d0 = p[(size_t)nb * 3 + 0] - p[(size_t)i * 3 + 0];
        const float d1 = p[(size_t)nb * 3 + 1] - p[(size_t)i * 3 + 1];
        const float d2 = p[(size_t)nb * 3 + 2] - p[(size_t)i * 3 + 2];
        #pragma unroll
        for (int c = 0; c < 3; ++c) {
            const float o = d0 * Wp1[0 * 3 + c] + d1 * Wp1[1 * 3 + c]
                          + d2 * Wp1[2 * 3 + c] + bp1[c];
            pr1[(size_t)pair * 3 + c] = o;
            atomicAdd(&sacc[c], o);
            atomicAdd(&sacc[4 + c], o * o);
        }
    }
    __syncthreads();
    if (threadIdx.x < 3) {
        atomicAdd(&S[128 + threadIdx.x], sacc[threadIdx.x]);
        atomicAdd(&S[136 + threadIdx.x], sacc[4 + threadIdx.x]);
    }
}

// bnp scale/shift coefficients (3 channels)
__device__ __forceinline__ void bnp_coeffs(
    const float* __restrict__ S, const float* __restrict__ g,
    const float* __restrict__ b, float invcnt, float psc[3], float psh[3])
{
    #pragma unroll
    for (int d = 0; d < 3; ++d) {
        const float m  = S[128 + d] * invcnt;
        const float v  = S[136 + d] * invcnt - m * m;
        const float rs = rsqrtf(v + EPSBN);
        psc[d] = rs * g[d];
        psh[d] = b[d] - m * rs * g[d];
    }
}

// pe channel c = relu(bnp(pr)) @ Wp2[:,c] + bp2[c]
__device__ __forceinline__ float pe_channel(
    const float pr[3], const float* __restrict__ Wp2, float bp2c,
    const float psc[3], const float psh[3], int c)
{
    float pe = bp2c;
    #pragma unroll
    for (int d = 0; d < 3; ++d) {
        const float t = fmaxf(fmaf(pr[d], psc[d], psh[d]), 0.0f);
        pe = fmaf(t, Wp2[d * 64 + c], pe);
    }
    return pe;
}

// ---------------------------------------------------------------------------
// bnw1 stats over virtual w0 = k[knn] - q + pe  (channels=64, count=N*NS)
// ---------------------------------------------------------------------------
__global__ __launch_bounds__(256) void w0_stats_k(
    const float* __restrict__ qb, const float* __restrict__ kb,
    const int* __restrict__ knn, const float* __restrict__ pr1,
    const float* __restrict__ Wp2, const float* __restrict__ bp2,
    const float* __restrict__ S, const float* __restrict__ bnp_g,
    const float* __restrict__ bnp_b, float* __restrict__ Sout,
    int npairs, float invcnt)
{
    __shared__ float ssum[256], ssq[256];
    const int c = threadIdx.x & 63;
    const int g = threadIdx.x >> 6;
    float psc[3], psh[3];
    bnp_coeffs(S, bnp_g, bnp_b, invcnt, psc, psh);
    const float bp2c = bp2[c];

    float s = 0.0f, q = 0.0f;
    for (int pair = blockIdx.x * 4 + g; pair < npairs; pair += gridDim.x * 4) {
        const int i  = pair >> 4;
        const int nb = knn[pair];
        float pr[3];
        #pragma unroll
        for (int d = 0; d < 3; ++d) pr[d] = pr1[(size_t)pair * 3 + d];
        const float pe = pe_channel(pr, Wp2, bp2c, psc, psh, c);
        const float w0 = kb[(size_t)nb * 64 + c] - qb[(size_t)i * 64 + c] + pe;
        s += w0; q += w0 * w0;
    }
    ssum[threadIdx.x] = s; ssq[threadIdx.x] = q;
    __syncthreads();
    if (threadIdx.x < 64) {
        s = ssum[c] + ssum[c + 64] + ssum[c + 128] + ssum[c + 192];
        q = ssq[c]  + ssq[c + 64]  + ssq[c + 128]  + ssq[c + 192];
        atomicAdd(&Sout[192 + c], s);
        atomicAdd(&Sout[256 + c], q);
    }
}

// ---------------------------------------------------------------------------
// w1[pair,0:8] = relu(bnw1(w0)) @ Ww1 + bw1 ; accumulate bnw2 stats.
// 256 threads = 4 groups of 64; each group handles one pair.
// ---------------------------------------------------------------------------
__global__ __launch_bounds__(256) void w1_stage_k(
    const float* __restrict__ qb, const float* __restrict__ kb,
    const int* __restrict__ knn, const float* __restrict__ pr1,
    const float* __restrict__ Wp2, const float* __restrict__ bp2,
    const float* __restrict__ S, const float* __restrict__ bnp_g,
    const float* __restrict__ bnp_b, const float* __restrict__ bnw1_g,
    const float* __restrict__ bnw1_b, const float* __restrict__ Ww1,
    const float* __restrict__ bw1, float* __restrict__ w1out,
    float* __restrict__ Sout, int npairs, float invcnt)
{
    __shared__ float w0s[4][64];
    __shared__ float sacc[16];
    if (threadIdx.x < 16) sacc[threadIdx.x] = 0.0f;
    __syncthreads();

    const int c    = threadIdx.x & 63;
    const int g    = threadIdx.x >> 6;
    const int pair = blockIdx.x * 4 + g;

    float psc[3], psh[3];
    bnp_coeffs(S, bnp_g, bnp_b, invcnt, psc, psh);
    // bnw1 per-channel affine
    const float m1  = S[192 + c] * invcnt;
    const float v1  = S[256 + c] * invcnt - m1 * m1;
    const float rs1 = rsqrtf(v1 + EPSBN);
    const float sc1 = rs1 * bnw1_g[c];
    const float sh1 = bnw1_b[c] - m1 * rs1 * bnw1_g[c];

    if (pair < npairs) {
        const int i  = pair >> 4;
        const int nb = knn[pair];
        float pr[3];
        #pragma unroll
        for (int d = 0; d < 3; ++d) pr[d] = pr1[(size_t)pair * 3 + d];
        const float pe = pe_channel(pr, Wp2, bp2[c], psc, psh, c);
        const float w0 = kb[(size_t)nb * 64 + c] - qb[(size_t)i * 64 + c] + pe;
        w0s[g][c] = fmaxf(fmaf(w0, sc1, sh1), 0.0f);
    }
    __syncthreads();
    if (pair < npairs && c < 8) {
        float acc = bw1[c];
        #pragma unroll 8
        for (int d = 0; d < 64; ++d) acc = fmaf(w0s[g][d], Ww1[d * 8 + c], acc);
        w1out[(size_t)pair * 8 + c] = acc;
        atomicAdd(&sacc[c], acc);
        atomicAdd(&sacc[8 + c], acc * acc);
    }
    __syncthreads();
    if (threadIdx.x < 16) atomicAdd(&Sout[320 + threadIdx.x], sacc[threadIdx.x]);
}

// ---------------------------------------------------------------------------
// Per point: ww = relu(bnw2(w1)) @ Ww2 + bw2 ; softmax over NS ;
// agg[i,c] = sum_j (v[knn]+pe)[c] * ww[j][c&7]   (share-planes S=8)
// 256 threads = 4 groups of 64 (one point each).
// ---------------------------------------------------------------------------
__global__ __launch_bounds__(256) void softmax_agg_k(
    const float* __restrict__ vb, const int* __restrict__ knn,
    const float* __restrict__ pr1, const float* __restrict__ w1,
    const float* __restrict__ Ww2, const float* __restrict__ bw2,
    const float* __restrict__ Wp2, const float* __restrict__ bp2,
    const float* __restrict__ S, const float* __restrict__ bnp_g,
    const float* __restrict__ bnp_b, const float* __restrict__ bnw2_g,
    const float* __restrict__ bnw2_b, float* __restrict__ agg,
    int npts, float invcnt)
{
    __shared__ float ww[4][16][8];
    const int c = threadIdx.x & 63;
    const int g = threadIdx.x >> 6;
    const int i = blockIdx.x * 4 + g;

    // bnw2 affine (8 channels, computed redundantly per thread — trivial)
    float sc2[8], sh2[8];
    #pragma unroll
    for (int d = 0; d < 8; ++d) {
        const float m  = S[320 + d] * invcnt;
        const float v  = S[328 + d] * invcnt - m * m;
        const float rs = rsqrtf(v + EPSBN);
        sc2[d] = rs * bnw2_g[d];
        sh2[d] = bnw2_b[d] - m * rs * bnw2_g[d];
    }

    if (i < npts) {
        for (int vv = c; vv < 128; vv += 64) {
            const int j = vv >> 3, c8 = vv & 7;
            float acc = bw2[c8];
            #pragma unroll
            for (int d = 0; d < 8; ++d) {
                float t = fmaf(w1[((size_t)i * 16 + j) * 8 + d], sc2[d], sh2[d]);
                t = fmaxf(t, 0.0f);
                acc = fmaf(t, Ww2[d * 8 + c8], acc);
            }
            ww[g][j][c8] = acc;
        }
    }
    __syncthreads();
    if (i < npts && c < 8) {
        float mx = -1e30f;
        #pragma unroll
        for (int j = 0; j < 16; ++j) mx = fmaxf(mx, ww[g][j][c]);
        float sum = 0.0f;
        #pragma unroll
        for (int j = 0; j < 16; ++j) {
            const float e = __expf(ww[g][j][c] - mx);
            ww[g][j][c] = e; sum += e;
        }
        const float inv = 1.0f / sum;
        #pragma unroll
        for (int j = 0; j < 16; ++j) ww[g][j][c] *= inv;
    }
    __syncthreads();
    if (i < npts) {
        float psc[3], psh[3];
        bnp_coeffs(S, bnp_g, bnp_b, invcnt, psc, psh);
        const float bp2c = bp2[c];
        const int c8 = c & 7;
        float acc = 0.0f;
        #pragma unroll 4
        for (int j = 0; j < 16; ++j) {
            const int pair = i * 16 + j;
            const int nb   = knn[pair];
            float pr[3];
            #pragma unroll
            for (int d = 0; d < 3; ++d) pr[d] = pr1[(size_t)pair * 3 + d];
            const float pe = pe_channel(pr, Wp2, bp2c, psc, psh, c);
            acc = fmaf(vb[(size_t)nb * 64 + c] + pe, ww[g][j][c8], acc);
        }
        agg[(size_t)i * 64 + c] = acc;
    }
}

// out = relu(bn3(u3) + x)
__global__ __launch_bounds__(256) void final_k(
    const float* __restrict__ u3, const float* __restrict__ x,
    const float* __restrict__ S, const float* __restrict__ g3,
    const float* __restrict__ b3, float* __restrict__ out,
    int total, float invN)
{
    const int i = blockIdx.x * blockDim.x + threadIdx.x;
    if (i >= total) return;
    const int c = i & 63;
    const float m = S[512 + c] * invN;
    const float v = S[576 + c] * invN - m * m;
    const float y = (u3[i] - m) * rsqrtf(v + EPSBN) * g3[c] + b3[c];
    out[i] = fmaxf(y + x[i], 0.0f);
}

// ---------------------------------------------------------------------------

extern "C" void kernel_launch(void* const* d_in, const int* in_sizes, int n_in,
                              void* d_out, int out_size, void* d_ws, size_t ws_size,
                              hipStream_t stream) {
    const float* p     = (const float*)d_in[0];
    const float* x     = (const float*)d_in[1];
    const int*   knn   = (const int*)  d_in[2];
    const float* W1    = (const float*)d_in[3];
    const float* bn1_g = (const float*)d_in[4];
    const float* bn1_b = (const float*)d_in[5];
    const float* Wq    = (const float*)d_in[6];
    const float* bq    = (const float*)d_in[7];
    const float* Wk    = (const float*)d_in[8];
    const float* bk    = (const float*)d_in[9];
    const float* Wv    = (const float*)d_in[10];
    const float* bv    = (const float*)d_in[11];
    const float* Wp1   = (const float*)d_in[12];
    const float* bp1   = (const float*)d_in[13];
    const float* bnp_g = (const float*)d_in[14];
    const float* bnp_b = (const float*)d_in[15];
    const float* Wp2   = (const float*)d_in[16];
    const float* bp2   = (const float*)d_in[17];
    const float* bnw1_g= (const float*)d_in[18];
    const float* bnw1_b= (const float*)d_in[19];
    const float* Ww1   = (const float*)d_in[20];
    const float* bw1   = (const float*)d_in[21];
    const float* bnw2_g= (const float*)d_in[22];
    const float* bnw2_b= (const float*)d_in[23];
    const float* Ww2   = (const float*)d_in[24];
    const float* bw2   = (const float*)d_in[25];
    const float* bn2_g = (const float*)d_in[26];
    const float* bn2_b = (const float*)d_in[27];
    const float* W3    = (const float*)d_in[28];
    const float* bn3_g = (const float*)d_in[29];
    const float* bn3_b = (const float*)d_in[30];

    const int N  = in_sizes[0] / 3;          // 80000
    const int NP = N * 16;                   // pairs (N*NS)
    const size_t NC = (size_t)N * 64;
    const float invN = 1.0f / (float)N;
    const float invP = 1.0f / (float)NP;

    // workspace carve-up
    float* S = (float*)d_ws;                              // 1024 floats
    _Float16* wf = (_Float16*)(S + STATS_FLOATS);         // 5 * 4096 halves
    _Float16* wfW1 = wf + 0 * WFRAG_HALVES;
    _Float16* wfWq = wf + 1 * WFRAG_HALVES;
    _Float16* wfWk = wf + 2 * WFRAG_HALVES;
    _Float16* wfWv = wf + 3 * WFRAG_HALVES;
    _Float16* wfW3 = wf + 4 * WFRAG_HALVES;
    float* u1  = S + STATS_FLOATS + (5 * WFRAG_HALVES) / 2;   // N*64
    float* y   = u1  + NC;                   // N*64
    float* qb  = y   + NC;                   // N*64
    float* kb  = qb  + NC;                   // N*64
    float* vb  = kb  + NC;                   // N*64
    float* pr1 = vb  + NC;                   // N*16*3
    float* w1b = pr1 + (size_t)N * 48;       // N*16*8
    float* agg = u1;                         // reuse (u1 dead after bn1 apply)
    float* y2  = y;                          // reuse (y dead after q/k/v gemms)
    float* u3  = qb;                         // reuse (qb dead after w1 stage)

    const dim3 B(256);
    const int gGemm = (N + 127) / 128;
    const int gElem = (int)((NC + 255) / 256);
    const int gPair = (NP + 255) / 256;
    const int gGrp4 = (NP + 3) / 4;
    const int gPt4  = (N + 3) / 4;
    const int gPrep = WFRAG_HALVES / 256;

    zero_stats_k<<<1, B, 0, stream>>>(S);
    prep_weight_k<<<gPrep, B, 0, stream>>>(W1, wfW1);
    prep_weight_k<<<gPrep, B, 0, stream>>>(Wq, wfWq);
    prep_weight_k<<<gPrep, B, 0, stream>>>(Wk, wfWk);
    prep_weight_k<<<gPrep, B, 0, stream>>>(Wv, wfWv);
    prep_weight_k<<<gPrep, B, 0, stream>>>(W3, wfW3);

    // y = relu(bn1(x @ W1))
    gemm64_wmma_k<<<gGemm, B, 0, stream>>>(x, wfW1, nullptr, u1, N);
    colstats64_k<<<512, B, 0, stream>>>(u1, S + 0, S + 64, N);
    bn_relu_k<<<gElem, B, 0, stream>>>(u1, S + 0, S + 64, bn1_g, bn1_b, y, (int)NC, invN);

    // q, k, v
    gemm64_wmma_k<<<gGemm, B, 0, stream>>>(y, wfWq, bq, qb, N);
    gemm64_wmma_k<<<gGemm, B, 0, stream>>>(y, wfWk, bk, kb, N);
    gemm64_wmma_k<<<gGemm, B, 0, stream>>>(y, wfWv, bv, vb, N);

    // positional encoding stage 1 + bnp stats
    pr_stage_k<<<gPair, B, 0, stream>>>(p, knn, Wp1, bp1, pr1, S, NP);

    // bnw1 stats over virtual w0
    w0_stats_k<<<4096, B, 0, stream>>>(qb, kb, knn, pr1, Wp2, bp2,
                                       S, bnp_g, bnp_b, S, NP, invP);

    // w1 = relu(bnw1(w0)) @ Ww1 + bw1 ; bnw2 stats
    w1_stage_k<<<gGrp4, B, 0, stream>>>(qb, kb, knn, pr1, Wp2, bp2,
                                        S, bnp_g, bnp_b, bnw1_g, bnw1_b,
                                        Ww1, bw1, w1b, S, NP, invP);

    // softmax over neighbors + share-planes aggregation
    softmax_agg_k<<<gPt4, B, 0, stream>>>(vb, knn, pr1, w1b, Ww2, bw2,
                                          Wp2, bp2, S, bnp_g, bnp_b,
                                          bnw2_g, bnw2_b, agg, N, invP);

    // y2 = relu(bn2(agg)) ; u3 = y2 @ W3 ; out = relu(bn3(u3) + x)
    colstats64_k<<<512, B, 0, stream>>>(agg, S + 384, S + 448, N);
    bn_relu_k<<<gElem, B, 0, stream>>>(agg, S + 384, S + 448, bn2_g, bn2_b, y2, (int)NC, invN);
    gemm64_wmma_k<<<gGemm, B, 0, stream>>>(y2, wfW3, nullptr, u3, N);
    colstats64_k<<<512, B, 0, stream>>>(u3, S + 512, S + 576, N);
    final_k<<<gElem, B, 0, stream>>>(u3, x, S, bn3_g, bn3_b, (float*)d_out, (int)NC, invN);
}
